// XMem_13091060318871
// MI455X (gfx1250) — matrix operand order repaired and checked by reading, not compiled, not run on hardware
//
#include <hip/hip_runtime.h>
#include <hip/hip_bf16.h>
#include <stdint.h>

// Problem dims (fixed by the reference)
#define CK_   64
#define CV_   512
#define NOBJ_ 2
#define T_    2
#define H_    64
#define W_    64
#define HW_   (H_*W_)       // 4096 query positions
#define N_    (T_*HW_)      // 8192 memory slots
#define V_    (NOBJ_*CV_)   // 1024 value channels
#define KC_   (2*CK_)       // 128 combined-K for the similarity GEMM

typedef __bf16 bf16_t;
typedef __attribute__((ext_vector_type(16))) __bf16 v16bf;
typedef __attribute__((ext_vector_type(8)))  __bf16 v8bf;
typedef __attribute__((ext_vector_type(8)))  float  v8f;

__device__ __forceinline__ unsigned short f32_to_bf16(float f) {
    union { float f; unsigned u; } v; v.f = f;
    unsigned u = v.u;
    if ((u & 0x7fffffffu) > 0x7f800000u) return (unsigned short)((u >> 16) | 0x0040u); // NaN
    unsigned r = u + 0x7fffu + ((u >> 16) & 1u);   // round-to-nearest-even
    return (unsigned short)(r >> 16);
}

// Low 32 bits of a generic pointer derived from LDS = wave-relative LDS offset
__device__ __forceinline__ unsigned lds_off(const void* p) {
    return (unsigned)(uintptr_t)p;
}

// ---- WMMA fragment loaders (ISA 7.12.2, wave32) ------------------------------
// A (M=16 x K=32, bf16, row-major, stride lda):
//   lanes 0-15 : row M=lane, K = k0+0..15
//   lanes 16-31: row M=lane-16, K = k0+16..31
__device__ __forceinline__ v16bf load_a_frag(const bf16_t* A, int lda, int row0,
                                             int k0, int lane) {
    int half = lane >> 4, m = lane & 15;
    const bf16_t* p = A + (size_t)(row0 + m) * lda + k0 + half * 16;
    v8bf lo = *(const v8bf*)(p);
    v8bf hi = *(const v8bf*)(p + 8);
    return __builtin_shufflevector(lo, hi, 0,1,2,3,4,5,6,7,8,9,10,11,12,13,14,15);
}
// B (K=32 x N=16, bf16) sourced from a TRANSPOSED store Bt (N-major, stride ldb):
//   lanes 0-15 : col N=lane,  K = k0+0..15   (contiguous in Bt row)
//   lanes 16-31: col N=lane-16, K = k0+16..31
__device__ __forceinline__ v16bf load_b_frag(const bf16_t* Bt, int ldb, int col0,
                                             int k0, int lane) {
    int half = lane >> 4, n = lane & 15;
    const bf16_t* p = Bt + (size_t)(col0 + n) * ldb + k0 + half * 16;
    return *(const v16bf*)(p);
}
// Same fragment shape read from an LDS slab laid out [64 cols][64 K] bf16
__device__ __forceinline__ v16bf load_b_lds(const bf16_t* Bl, int colGrp, int k0, int lane) {
    int half = lane >> 4, n = lane & 15;
    return *(const v16bf*)(Bl + (colGrp * 16 + n) * 64 + k0 + half * 16);
}

// ---- Kernel 1: build Bt_comb = [qk*qe ; qe]^T (HW x 128) and bsq -------------
__global__ void xmem_prep_q(const float* __restrict__ qk, const float* __restrict__ qe,
                            unsigned short* __restrict__ Bt, float* __restrict__ bsq) {
    int p = blockIdx.x * blockDim.x + threadIdx.x;
    if (p >= HW_) return;
    float acc = 0.f;
    #pragma unroll 4
    for (int c = 0; c < CK_; ++c) {
        float k = qk[(size_t)c * HW_ + p];
        float e = qe[(size_t)c * HW_ + p];
        acc += e * k * k;
        Bt[(size_t)p * KC_ + c]       = f32_to_bf16(k * e);
        Bt[(size_t)p * KC_ + CK_ + c] = f32_to_bf16(e);
    }
    bsq[p] = acc;
}

// ---- Kernel 2: build A_comb = [2*mk ; -mk^2]^T (N x 128) ---------------------
__global__ void xmem_prep_mk(const float* __restrict__ mk, unsigned short* __restrict__ Ac) {
    int n = blockIdx.x * blockDim.x + threadIdx.x;
    if (n >= N_) return;
    #pragma unroll 4
    for (int c = 0; c < CK_; ++c) {
        float k = mk[(size_t)c * N_ + n];
        Ac[(size_t)n * KC_ + c]       = f32_to_bf16(2.0f * k);
        Ac[(size_t)n * KC_ + CK_ + c] = f32_to_bf16(-k * k);
    }
}

// ---- Kernel 3: memory_value fp32 -> bf16 (V x N, row-major, K contiguous) ----
__global__ void xmem_conv_mv(const float* __restrict__ mv, unsigned short* __restrict__ mvb) {
    size_t i = ((size_t)blockIdx.x * blockDim.x + threadIdx.x) * 4;
    if (i >= (size_t)V_ * N_) return;
    float4 f = *(const float4*)(mv + i);
    unsigned lo = (unsigned)f32_to_bf16(f.x) | ((unsigned)f32_to_bf16(f.y) << 16);
    unsigned hi = (unsigned)f32_to_bf16(f.z) | ((unsigned)f32_to_bf16(f.w) << 16);
    uint2 u; u.x = lo; u.y = hi;
    *(uint2*)(mvb + i) = u;
}

// ---- Kernel 4: simT(p,n) = (A_comb x B_comb - bsq[p]) * ms[n]/sqrt(CK) -------
// One wave computes a 16(n) x 64(p) tile; K = 128 in 4 WMMA steps; A reused x4.
__global__ void xmem_gemm_sim(const bf16_t* __restrict__ Ac, const bf16_t* __restrict__ Bt,
                              const float* __restrict__ bsq, const float* __restrict__ ms,
                              float* __restrict__ simT) {
    int lane = threadIdx.x & 31;
    int w = (int)((blockIdx.x * blockDim.x + threadIdx.x) >> 5);
    int n0 = (w & (N_ / 16 - 1)) * 16;       // 512 row tiles
    int p0 = (w >> 9) * 64;                  // 64 col groups
    v8f z = {0.f,0.f,0.f,0.f,0.f,0.f,0.f,0.f};
    v8f acc0 = z, acc1 = z, acc2 = z, acc3 = z;
    #pragma unroll
    for (int kk = 0; kk < KC_; kk += 32) {
        v16bf a  = load_a_frag(Ac, KC_, n0, kk, lane);
        v16bf b0 = load_b_frag(Bt, KC_, p0 +  0, kk, lane);
        v16bf b1 = load_b_frag(Bt, KC_, p0 + 16, kk, lane);
        v16bf b2 = load_b_frag(Bt, KC_, p0 + 32, kk, lane);
        v16bf b3 = load_b_frag(Bt, KC_, p0 + 48, kk, lane);
        acc0 = __builtin_amdgcn_wmma_f32_16x16x32_bf16(false, a, false, b0, (short)0, acc0, false, false);
        acc1 = __builtin_amdgcn_wmma_f32_16x16x32_bf16(false, a, false, b1, (short)0, acc1, false, false);
        acc2 = __builtin_amdgcn_wmma_f32_16x16x32_bf16(false, a, false, b2, (short)0, acc2, false, false);
        acc3 = __builtin_amdgcn_wmma_f32_16x16x32_bf16(false, a, false, b3, (short)0, acc3, false, false);
    }
    // Epilogue: C element r in lane -> row n = n0 + r + 8*half, col p = p0 + 16j + (lane&15)
    const float inv = 0.125f;                // 1/sqrt(64)
    int half = lane >> 4, q = lane & 15;
    const float* msp = ms + n0 + half * 8;
    float msv[8];
    #pragma unroll
    for (int r = 0; r < 8; ++r) msv[r] = msp[r] * inv;
    v8f accs[4] = {acc0, acc1, acc2, acc3};
    #pragma unroll
    for (int j = 0; j < 4; ++j) {
        int p = p0 + 16 * j + q;
        float bq = bsq[p];
        v8f out;
        #pragma unroll
        for (int r = 0; r < 8; ++r) out[r] = (accs[j][r] - bq) * msv[r];
        *(v8f*)(simT + (size_t)p * N_ + n0 + half * 8) = out;   // contiguous 32B store
    }
}

// ---- Kernel 5: per-column softmax over N; overwrite simT row with bf16 P -----
// The 32 KB row is pulled into LDS with GLOBAL_LOAD_ASYNC_TO_LDS_B128 (ASYNCcnt),
// avoiding the VGPR round-trip entirely.
__global__ void xmem_softmax(float* __restrict__ simT, float* __restrict__ rsum) {
    __shared__ __align__(64) float srow[N_];     // 32 KB (WGP has 320 KB LDS)
    __shared__ float sred[256];
    int p = blockIdx.x;
    int t = threadIdx.x;
    float* row = simT + (size_t)p * N_;

    // async copy: 8 x 16B per thread, stride 4 KB (offset: applies to LDS and mem)
    {
        unsigned long long ga = (unsigned long long)(uintptr_t)(row + t * 4);
        unsigned ld = lds_off(&srow[t * 4]);
        asm volatile(
            "global_load_async_to_lds_b128 %0, %1, off\n\t"
            "global_load_async_to_lds_b128 %0, %1, off offset:4096\n\t"
            "global_load_async_to_lds_b128 %0, %1, off offset:8192\n\t"
            "global_load_async_to_lds_b128 %0, %1, off offset:12288\n\t"
            "global_load_async_to_lds_b128 %0, %1, off offset:16384\n\t"
            "global_load_async_to_lds_b128 %0, %1, off offset:20480\n\t"
            "global_load_async_to_lds_b128 %0, %1, off offset:24576\n\t"
            "global_load_async_to_lds_b128 %0, %1, off offset:28672"
            :: "v"(ld), "v"(ga) : "memory");
        asm volatile("s_wait_asynccnt 0x0" ::: "memory");
    }
    __syncthreads();

    float m = -3.4e38f;
    for (int i = t; i < N_; i += 256) m = fmaxf(m, srow[i]);
    sred[t] = m; __syncthreads();
    for (int s = 128; s > 0; s >>= 1) { if (t < s) sred[t] = fmaxf(sred[t], sred[t + s]); __syncthreads(); }
    float mx = sred[0]; __syncthreads();
    // whole row is staged in LDS -> safe to overwrite global row with bf16 exp values
    unsigned short* prow = (unsigned short*)row;   // Pt row p, pitch 2*N_ bf16 elems
    float sum = 0.f;
    for (int i = t; i < N_; i += 256) {
        float e = __expf(srow[i] - mx);
        sum += e;
        prow[i] = f32_to_bf16(e);
    }
    sred[t] = sum; __syncthreads();
    for (int s = 128; s > 0; s >>= 1) { if (t < s) sred[t] += sred[t + s]; __syncthreads(); }
    if (t == 0) rsum[p] = 1.0f / sred[0];
}

// ---- Kernel 6: out(v,p) = rsum[p] * sum_n mv_bf16(v,n) * P(n,p) --------------
// Block = 8 waves covering 128(v) x 64(p). B K-slabs (64 wide, 8 KB) are staged
// into LDS with async global->LDS copies (double-buffered); each wave owns a
// private 16-row A tile streamed from L2 and reads B fragments from LDS.
#define KSLAB_ 64
__global__ void xmem_gemm_out(const bf16_t* __restrict__ mvb, const bf16_t* __restrict__ Pt,
                              const float* __restrict__ rsum, float* __restrict__ out) {
    __shared__ __align__(64) unsigned short Bslab[2][64 * KSLAB_];   // 2 x 8 KB

    const int tid  = threadIdx.x;
    const int lane = tid & 31;
    const int wid  = tid >> 5;                       // 0..7
    const int p0   = (blockIdx.x >> 3) * 64;         // 64 col groups
    const int v0   = (blockIdx.x & 7) * 128 + wid * 16;
    const int ldb  = 2 * N_;                         // Pt pitch in bf16 elems

    // cooperative stage geometry: 256 threads x one 32B chunk = 8 KB slab
    const int scol = tid >> 2;                       // 0..63
    const int sko  = (tid & 3) * 16;                 // 0,16,32,48
    const bf16_t* gsrc = Pt + (size_t)(p0 + scol) * ldb + sko;
    const unsigned ldsA = lds_off(&Bslab[0][scol * KSLAB_ + sko]);
    const unsigned ldsB = lds_off(&Bslab[1][scol * KSLAB_ + sko]);

    // prologue: stage slab 0 into buffer 0 (memory -> LDS direct, ASYNCcnt)
    {
        unsigned long long ga = (unsigned long long)(uintptr_t)gsrc;
        asm volatile(
            "global_load_async_to_lds_b128 %0, %1, off\n\t"
            "global_load_async_to_lds_b128 %0, %1, off offset:16"
            :: "v"(ldsA), "v"(ga) : "memory");
    }

    v8f z = {0.f,0.f,0.f,0.f,0.f,0.f,0.f,0.f};
    v8f acc0 = z, acc1 = z, acc2 = z, acc3 = z;

    const int NSTAGE = N_ / KSLAB_;                  // 128
    for (int s = 0; s < NSTAGE; ++s) {
        asm volatile("s_wait_asynccnt 0x0" ::: "memory");
        __syncthreads();                             // slab s landed in LDS; prev buffer free
        if (s + 1 < NSTAGE) {                        // stage s+1 into the other buffer
            unsigned long long ga =
                (unsigned long long)(uintptr_t)(gsrc + (size_t)(s + 1) * KSLAB_);
            unsigned ld = ((s + 1) & 1) ? ldsB : ldsA;
            asm volatile(
                "global_load_async_to_lds_b128 %0, %1, off\n\t"
                "global_load_async_to_lds_b128 %0, %1, off offset:16"
                :: "v"(ld), "v"(ga) : "memory");
            // pull next A slab toward the near cache (global_prefetch_b8)
            __builtin_prefetch(mvb + (size_t)(v0 + (lane & 15)) * N_ + (s + 1) * KSLAB_, 0, 0);
        }
        const bf16_t* Bl = (const bf16_t*)Bslab[s & 1];
        #pragma unroll
        for (int kh = 0; kh < 2; ++kh) {
            int kk = s * KSLAB_ + kh * 32;
            v16bf a  = load_a_frag(mvb, N_, v0, kk, lane);
            v16bf b0 = load_b_lds(Bl, 0, kh * 32, lane);
            v16bf b1 = load_b_lds(Bl, 1, kh * 32, lane);
            v16bf b2 = load_b_lds(Bl, 2, kh * 32, lane);
            v16bf b3 = load_b_lds(Bl, 3, kh * 32, lane);
            acc0 = __builtin_amdgcn_wmma_f32_16x16x32_bf16(false, a, false, b0, (short)0, acc0, false, false);
            acc1 = __builtin_amdgcn_wmma_f32_16x16x32_bf16(false, a, false, b1, (short)0, acc1, false, false);
            acc2 = __builtin_amdgcn_wmma_f32_16x16x32_bf16(false, a, false, b2, (short)0, acc2, false, false);
            acc3 = __builtin_amdgcn_wmma_f32_16x16x32_bf16(false, a, false, b3, (short)0, acc3, false, false);
        }
    }

    int half = lane >> 4, q = lane & 15;
    v8f accs[4] = {acc0, acc1, acc2, acc3};
    #pragma unroll
    for (int j = 0; j < 4; ++j) {
        int p = p0 + 16 * j + q;
        float rs = rsum[p];
        #pragma unroll
        for (int r = 0; r < 8; ++r)
            out[(size_t)(v0 + half * 8 + r) * HW_ + p] = accs[j][r] * rs;
    }
}

extern "C" void kernel_launch(void* const* d_in, const int* in_sizes, int n_in,
                              void* d_out, int out_size, void* d_ws, size_t ws_size,
                              hipStream_t stream) {
    const float* qk = (const float*)d_in[0];   // query_key        (CK, HW)
    const float* qe = (const float*)d_in[1];   // query_selection  (CK, HW)
    const float* mk = (const float*)d_in[2];   // memory_key       (CK, N)
    const float* ms = (const float*)d_in[3];   // memory_shrinkage (N)
    const float* mv = (const float*)d_in[4];   // memory_value     (V, N)
    float* out = (float*)d_out;                // (V, HW) fp32

    char* ws = (char*)d_ws;
    size_t o = 0;
    unsigned short* Ac  = (unsigned short*)(ws + o); o += (size_t)N_  * KC_ * 2; // 2 MB
    unsigned short* Bt  = (unsigned short*)(ws + o); o += (size_t)HW_ * KC_ * 2; // 1 MB
    float* bsq          = (float*)(ws + o);          o += (size_t)HW_ * 4;
    float* rsum         = (float*)(ws + o);          o += (size_t)HW_ * 4;
    o = (o + 255) & ~(size_t)255;
    unsigned short* mvb = (unsigned short*)(ws + o); o += (size_t)V_ * N_ * 2;   // 16 MB
    o = (o + 255) & ~(size_t)255;
    float* simT         = (float*)(ws + o);          o += (size_t)HW_ * N_ * 4;  // 128 MB (reused as Pt bf16)

    // prep
    xmem_prep_q <<<HW_ / 256, 256, 0, stream>>>(qk, qe, Bt, bsq);
    xmem_prep_mk<<<N_  / 256, 256, 0, stream>>>(mk, Ac);
    xmem_conv_mv<<<((size_t)V_ * N_ / 4) / 256, 256, 0, stream>>>(mv, mvb);

    // similarity GEMM: 512 n-tiles * 64 p-groups = 32768 waves -> 4096 blocks of 8 waves
    xmem_gemm_sim<<<(N_ / 16) * (HW_ / 64) / 8, 256, 0, stream>>>(
        (const bf16_t*)Ac, (const bf16_t*)Bt, bsq, ms, simT);

    // softmax over memory slots, one block per query column
    xmem_softmax<<<HW_, 256, 0, stream>>>(simT, rsum);

    // readout GEMM: (V/128) row blocks * (HW/64) col groups = 512 blocks of 8 waves
    xmem_gemm_out<<<(V_ / 128) * (HW_ / 64), 256, 0, stream>>>(
        (const bf16_t*)mvb, (const bf16_t*)simT, rsum, out);
}
